// STGNNI_16569983828529
// MI455X (gfx1250) — matrix operand order repaired and checked by reading, not compiled, not run on hardware
//
#include <hip/hip_runtime.h>
#include <hip/hip_bf16.h>
#include <math.h>

// ---------------------------------------------------------------------------
// Types for WMMA (CDNA5 gfx1250, wave32)
// ---------------------------------------------------------------------------
typedef __attribute__((ext_vector_type(16))) __bf16        v16bf;
typedef __attribute__((ext_vector_type(8)))  float         v8f;
typedef __attribute__((ext_vector_type(4)))  unsigned int  u32x4;

union BFrag {
    u32x4          q[2];
    unsigned short u[16];
    v16bf          bf;
};

static __device__ __forceinline__ unsigned short f2bf(float f) {
    unsigned int u = __builtin_bit_cast(unsigned int, f);
    unsigned int r = u + 0x7FFFu + ((u >> 16) & 1u);   // round-to-nearest-even
    return (unsigned short)(r >> 16);
}

// ---------------------------------------------------------------------------
// Problem constants
// ---------------------------------------------------------------------------
#define NN    2048
#define TT    1000
#define BB    32
#define C1    32
#define C2    64
#define C3    128
#define T2    994        // conv2 valid length
#define W2    248        // pooled length after conv2
#define T3    242        // conv3 valid length
#define W3    60         // pooled length after conv3
#define FLAT  (C3 * W3)  // 7680
#define EMB   768
#define NCLS  4
#define EPSV  1e-5f
#define KPASS 224        // K-chunk per conv staging pass (= 32ci * 7k)

// ---------------------------------------------------------------------------
// Kernel 1: conv1 batch statistics (sum / sumsq per channel)
// ---------------------------------------------------------------------------
__global__ __launch_bounds__(256) void conv1_stats_k(
    const float* __restrict__ x, const float* __restrict__ w,
    const float* __restrict__ b, float* __restrict__ stats)
{
    __shared__ float red[64];
    int tid = threadIdx.x;
    if (tid < 64) red[tid] = 0.f;
    __syncthreads();
    int n = blockIdx.x >> 4;
    int chunk = blockIdx.x & 15;
    int c = tid & 31;
    int j = tid >> 5;
    float wc[7];
#pragma unroll
    for (int k = 0; k < 7; ++k) wc[k] = w[c * 7 + k];
    float bc = b[c];
    const float* xn = x + (size_t)n * TT;
    float s = 0.f, q = 0.f;
    for (int i = 0; i < 8; ++i) {
        int t = chunk * 64 + j * 8 + i;
        if (t < TT) {
            float y = bc;
#pragma unroll
            for (int k = 0; k < 7; ++k) {
                int tt = t + k - 3;
                float xv = (tt >= 0 && tt < TT) ? xn[tt] : 0.f;
                y = fmaf(wc[k], xv, y);
            }
            s += y; q += y * y;
        }
    }
    atomicAdd(&red[c], s);
    atomicAdd(&red[32 + c], q);
    __syncthreads();
    if (tid < 32) {
        atomicAdd(&stats[tid], red[tid]);
        atomicAdd(&stats[32 + tid], red[32 + tid]);
    }
}

// ---------------------------------------------------------------------------
// Kernel 2: finalize BN -> per-channel scale/shift
// ---------------------------------------------------------------------------
__global__ void finalize_bn_k(const float* __restrict__ stats,
                              const float* __restrict__ g,
                              const float* __restrict__ bt,
                              float invcount, int C,
                              float* __restrict__ sc, float* __restrict__ sh)
{
    int c = threadIdx.x;
    if (c < C) {
        float mean = stats[c] * invcount;
        float var  = stats[C + c] * invcount - mean * mean;
        float s = g[c] * rsqrtf(var + EPSV);
        sc[c] = s;
        sh[c] = bt[c] - mean * s;
    }
}

// ---------------------------------------------------------------------------
// Kernel 3: conv1 recompute + BN + ReLU -> h1 bf16 [n][c][t]
// ---------------------------------------------------------------------------
__global__ __launch_bounds__(256) void conv1_apply_k(
    const float* __restrict__ x, const float* __restrict__ w,
    const float* __restrict__ b, const float* __restrict__ sc,
    const float* __restrict__ sh, unsigned short* __restrict__ out)
{
    size_t idx = (size_t)blockIdx.x * 256 + threadIdx.x;   // N*32*1000 exact
    int t = (int)(idx % TT);
    int c = (int)((idx / TT) % C1);
    size_t n = idx / (C1 * TT);
    const float* xn = x + n * TT;
    float y = b[c];
#pragma unroll
    for (int k = 0; k < 7; ++k) {
        int tt = t + k - 3;
        float xv = (tt >= 0 && tt < TT) ? xn[tt] : 0.f;
        y = fmaf(w[c * 7 + k], xv, y);
    }
    y = fmaxf(sc[c] * y + sh[c], 0.f);
    out[idx] = f2bf(y);
}

// ---------------------------------------------------------------------------
// Kernel 4: fused conv (WMMA GEMM) + bias + batch-stats + avg-pool/4
// im2col expansion done ONCE at staging time into sAex[m=64][kk] so the
// per-WMMA A fragment is two ds_load_b128 (no per-element /7, no scalar
// gathers in the hot loop).  K staged in passes of 224 (conv2: 1 pass,
// conv3: 2 passes).  LDS: 28KB (A) + 14KB (B) + 256B (stats).
// Block = 256 thr = 8 waves; per block: one n, 64 t-positions, 32 co.
// ---------------------------------------------------------------------------
__global__ __launch_bounds__(256) void conv_gemm_k(
    const unsigned short* __restrict__ hin, const float* __restrict__ w,
    const float* __restrict__ bias, float* __restrict__ pooled,
    float* __restrict__ stats,
    int CI, int Tin, int Tvalid, int CO, int CIK, int Wout)
{
    __shared__ __align__(16) unsigned short sAex[64 * KPASS];  // [m][kk_local]
    __shared__ __align__(16) unsigned short sB[32 * KPASS];    // [co][kk_local]
    __shared__ float red[64];

    const int tid = threadIdx.x;
    const int n   = blockIdx.z;
    const int t0  = blockIdx.y * 64;
    const int co0 = blockIdx.x * 32;

    if (tid < 64) red[tid] = 0.f;

    const unsigned short* hn = hin + (size_t)n * CI * Tin;
    const int lane = tid & 31;
    const int wv   = tid >> 5;
    const int wx   = wv & 1;          // co 16-tile
    const int wy   = wv >> 1;         // t  16-tile
    const int r16  = lane & 15;
    const int hf   = lane >> 4;

    v8f acc = {};
    const int nPass = CIK / KPASS;    // 1 (conv2) or 2 (conv3)
    for (int pass = 0; pass < nPass; ++pass) {
        const int base = pass * KPASS;
        // ---- stage expanded A: sAex[t][kkl] = h[ci][t0+t+k], kk=base+kkl ----
        for (int idx = tid; idx < 64 * KPASS; idx += 256) {
            int t   = idx & 63;
            int kkl = idx >> 6;
            int kkg = base + kkl;
            int ci  = kkg / 7;
            int k   = kkg - ci * 7;
            int gt  = t0 + t + k;
            sAex[t * KPASS + kkl] =
                (gt < Tin) ? hn[(size_t)ci * Tin + gt] : (unsigned short)0;
        }
        // ---- stage B: sB[co][kkl] = w[co0+co][base+kkl] (f32 -> bf16) ----
        for (int idx = tid; idx < 32 * KPASS; idx += 256) {
            int co = idx / KPASS, kkl = idx - co * KPASS;
            sB[co * KPASS + kkl] = f2bf(w[(size_t)(co0 + co) * CIK + base + kkl]);
        }
        __syncthreads();

        const unsigned short* arow = sAex + (wy * 16 + r16) * KPASS;
        const unsigned short* brow = sB   + (wx * 16 + r16) * KPASS;
#pragma unroll
        for (int kc = 0; kc < KPASS / 32; ++kc) {
            BFrag a, bm;
            const unsigned short* ap = arow + kc * 32 + 8 * hf;
            a.q[0] = *(const u32x4*)(ap);
            a.q[1] = *(const u32x4*)(ap + 16);
            const unsigned short* bp = brow + kc * 32 + 8 * hf;
            bm.q[0] = *(const u32x4*)(bp);
            bm.q[1] = *(const u32x4*)(bp + 16);
            acc = __builtin_amdgcn_wmma_f32_16x16x32_bf16(
                false, a.bf, false, bm.bf, (short)0, acc, false, false);
        }
        __syncthreads();
    }

    // ---- epilogue: bias, stats, pool ----
    const int co  = co0 + wx * 16 + r16;
    const float bco = bias[co];
    const int tbase = t0 + wy * 16 + hf * 8;
    float vals[8];
    float s = 0.f, q = 0.f;
#pragma unroll
    for (int rr = 0; rr < 8; ++rr) {
        float v = acc[rr] + bco;
        vals[rr] = v;
        if (tbase + rr < Tvalid) { s += v; q += v * v; }
    }
    atomicAdd(&red[wx * 16 + r16], s);
    atomicAdd(&red[32 + wx * 16 + r16], q);

    int wbase = tbase >> 2;
    float p0 = (vals[0] + vals[1] + vals[2] + vals[3]) * 0.25f;
    float p1 = (vals[4] + vals[5] + vals[6] + vals[7]) * 0.25f;
    float* prow = pooled + ((size_t)n * CO + co) * Wout;
    if (wbase < Wout)     prow[wbase]     = p0;
    if (wbase + 1 < Wout) prow[wbase + 1] = p1;

    __syncthreads();
    if (tid < 32) {
        atomicAdd(&stats[co0 + tid],      red[tid]);
        atomicAdd(&stats[CO + co0 + tid], red[32 + tid]);
    }
}

// ---------------------------------------------------------------------------
// Kernel 5: BN + ReLU on pooled f32 -> bf16
// ---------------------------------------------------------------------------
__global__ __launch_bounds__(256) void bn_relu_bf16_k(
    const float* __restrict__ in, const float* __restrict__ sc,
    const float* __restrict__ sh, unsigned short* __restrict__ out,
    int Wout, int CO)
{
    size_t idx = (size_t)blockIdx.x * 256 + threadIdx.x;  // exact multiple
    int c = (int)((idx / Wout) % CO);
    float v = fmaxf(sc[c] * in[idx] + sh[c], 0.f);
    out[idx] = f2bf(v);
}

// ---------------------------------------------------------------------------
// Kernel 6: position MLP  pe = relu(pos@W1+b1)@W2+b2   (one block per node)
// ---------------------------------------------------------------------------
__global__ __launch_bounds__(128) void pos_mlp_k(
    const float* __restrict__ pos, const float* __restrict__ w1,
    const float* __restrict__ b1, const float* __restrict__ w2,
    const float* __restrict__ b2, float* __restrict__ pe)
{
    __shared__ float hid[C3];
    int n = blockIdx.x, j = threadIdx.x;
    float p0 = pos[n * 3 + 0], p1 = pos[n * 3 + 1], p2 = pos[n * 3 + 2];
    float h = fmaf(p0, w1[j], fmaf(p1, w1[C3 + j], fmaf(p2, w1[2 * C3 + j], b1[j])));
    hid[j] = fmaxf(h, 0.f);
    __syncthreads();
    float acc = b2[j];
    for (int i = 0; i < C3; ++i) acc = fmaf(hid[i], w2[i * C3 + j], acc);
    pe[(size_t)n * C3 + j] = acc;
}

// ---------------------------------------------------------------------------
// Kernel 7: BN3 + ReLU + add pe -> flattened bf16 node features [n][c*60+w]
// ---------------------------------------------------------------------------
__global__ __launch_bounds__(256) void build_flat_k(
    const float* __restrict__ pooled3, const float* __restrict__ sc,
    const float* __restrict__ sh, const float* __restrict__ pe,
    unsigned short* __restrict__ out)
{
    size_t idx = (size_t)blockIdx.x * 256 + threadIdx.x;   // N*FLAT exact
    int c = (int)((idx / W3) % C3);
    size_t n = idx / FLAT;
    float v = fmaxf(sc[c] * pooled3[idx] + sh[c], 0.f) + pe[n * C3 + c];
    out[idx] = f2bf(v);
}

// ---------------------------------------------------------------------------
// Kernel 8: WMMA GEMM  C(f32, MxN) = A(bf16, MxK) * B(f32->bf16, KxN)
// 64x64 block tile, 8 waves, 2 accumulators/wave, K staged 128 at a time
// -> 8 WMMAs per wave per barrier pair.  LDS: 16KB A + 16KB Bt.
// Requires: M % 64 == 0, N % 64 == 0, K % 128 == 0.
// ---------------------------------------------------------------------------
__global__ __launch_bounds__(256) void gemm_bf16_k(
    const unsigned short* __restrict__ A, const float* __restrict__ Bw,
    float* __restrict__ C, int M, int Nn, int K)
{
    __shared__ __align__(16) unsigned short sA[64 * 128];   // [m][k]
    __shared__ __align__(16) unsigned short sBt[64 * 128];  // [n][k] transposed
    const int tid  = threadIdx.x;
    const int m0   = blockIdx.y * 64;
    const int n0   = blockIdx.x * 64;
    const int lane = tid & 31, wv = tid >> 5;
    const int wy   = wv >> 1;          // M 16-tile (0..3)
    const int wxp  = (wv & 1) * 2;     // first of two N 16-tiles
    const int r16  = lane & 15, hf = lane >> 4;

    v8f acc0 = {}, acc1 = {};
    for (int kc = 0; kc < K; kc += 128) {
        // stage A: 64x128 halves, 8 halves (16B) per thread x 4
#pragma unroll
        for (int i = 0; i < 4; ++i) {
            int c2   = tid + i * 256;           // 0..1023 chunks of 8 halves
            int row  = c2 >> 4;                 // 16 chunks per row
            int col8 = (c2 & 15) * 8;
            *(u32x4*)(sA + row * 128 + col8) =
                *(const u32x4*)(A + (size_t)(m0 + row) * K + kc + col8);
        }
        // stage B transposed: 128(K) x 64(N) f32 -> sBt[n][k] bf16
#pragma unroll
        for (int i = 0; i < 32; ++i) {
            int e  = tid + i * 256;             // 0..8191
            int nc = e & 63, kr = e >> 6;
            sBt[nc * 128 + kr] = f2bf(Bw[(size_t)(kc + kr) * Nn + n0 + nc]);
        }
        __syncthreads();
        const unsigned short* arow = sA  + (wy * 16 + r16) * 128;
        const unsigned short* br0  = sBt + (wxp * 16 + r16) * 128;
        const unsigned short* br1  = br0 + 16 * 128;
#pragma unroll
        for (int kcc = 0; kcc < 4; ++kcc) {
            BFrag a, b0, b1;
            const unsigned short* ap = arow + kcc * 32 + 8 * hf;
            a.q[0] = *(const u32x4*)(ap);
            a.q[1] = *(const u32x4*)(ap + 16);
            const unsigned short* bp0 = br0 + kcc * 32 + 8 * hf;
            b0.q[0] = *(const u32x4*)(bp0);
            b0.q[1] = *(const u32x4*)(bp0 + 16);
            const unsigned short* bp1 = br1 + kcc * 32 + 8 * hf;
            b1.q[0] = *(const u32x4*)(bp1);
            b1.q[1] = *(const u32x4*)(bp1 + 16);
            acc0 = __builtin_amdgcn_wmma_f32_16x16x32_bf16(
                false, a.bf, false, b0.bf, (short)0, acc0, false, false);
            acc1 = __builtin_amdgcn_wmma_f32_16x16x32_bf16(
                false, a.bf, false, b1.bf, (short)0, acc1, false, false);
        }
        __syncthreads();
    }
    int col0 = n0 + wxp * 16 + r16;
    int mrow = m0 + wy * 16 + hf * 8;
#pragma unroll
    for (int rr = 0; rr < 8; ++rr) {
        C[(size_t)(mrow + rr) * Nn + col0]      = acc0[rr];
        C[(size_t)(mrow + rr) * Nn + col0 + 16] = acc1[rr];
    }
}

// ---------------------------------------------------------------------------
// GCN support kernels
// ---------------------------------------------------------------------------
__global__ void deg_count_k(const int* __restrict__ dst, float* __restrict__ deg, int E)
{
    int e = blockIdx.x * 256 + threadIdx.x;
    if (e < E) atomicAdd(&deg[dst[e]], 1.0f);
}

__global__ __launch_bounds__(256) void edge_scatter_k(
    const float* __restrict__ g, const float* __restrict__ deg,
    const int* __restrict__ src, const int* __restrict__ dst,
    float* __restrict__ agg)
{
    int e = blockIdx.x;
    int s = src[e], d = dst[e];
    float coeff = rsqrtf(deg[s] + 1.f) * rsqrtf(deg[d] + 1.f);
    const float* gs = g + (size_t)s * EMB;
    float* ad = agg + (size_t)d * EMB;
    for (int f = threadIdx.x; f < EMB; f += 256)
        atomicAdd(&ad[f], gs[f] * coeff);
}

__global__ __launch_bounds__(256) void gcn_combine_k(
    const float* __restrict__ g, const float* __restrict__ agg,
    const float* __restrict__ deg, const float* __restrict__ bias,
    float* __restrict__ of32, unsigned short* __restrict__ obf)
{
    size_t idx = (size_t)blockIdx.x * 256 + threadIdx.x;  // N*EMB exact
    int f = (int)(idx % EMB);
    size_t i = idx / EMB;
    float v = agg[idx] + g[idx] / (deg[i] + 1.f) + bias[f];
    v = fmaxf(v, 0.f);
    if (of32) of32[idx] = v;
    if (obf)  obf[idx]  = f2bf(v);
}

__global__ void batch_count_k(const int* __restrict__ batch, float* __restrict__ cnt, int n)
{
    int i = blockIdx.x * 256 + threadIdx.x;
    if (i < n) atomicAdd(&cnt[batch[i]], 1.0f);
}

__global__ __launch_bounds__(256) void batch_pool_k(
    const float* __restrict__ h, const int* __restrict__ batch,
    float* __restrict__ pooled)
{
    size_t idx = (size_t)blockIdx.x * 256 + threadIdx.x;  // N*EMB exact
    int f = (int)(idx % EMB);
    size_t i = idx / EMB;
    atomicAdd(&pooled[(size_t)batch[i] * EMB + f], h[idx]);
}

__global__ __launch_bounds__(128) void final_dense_k(
    const float* __restrict__ pooled, const float* __restrict__ cnt,
    const float* __restrict__ dw, const float* __restrict__ db,
    float* __restrict__ out)
{
    __shared__ float lg[BB * NCLS];
    int tid = threadIdx.x;
    int bb = tid >> 2, k = tid & 3;
    float c = fmaxf(cnt[bb], 1.f);
    const float* pr = pooled + (size_t)bb * EMB;
    float acc = 0.f;
    for (int f = 0; f < EMB; ++f) acc = fmaf(pr[f], dw[f * NCLS + k], acc);
    acc = acc / c + db[k];
    lg[tid] = acc;
    __syncthreads();
    int base = bb * 4;
    float m = lg[base];
#pragma unroll
    for (int i = 1; i < 4; ++i) m = fmaxf(m, lg[base + i]);
    float se = 0.f;
#pragma unroll
    for (int i = 0; i < 4; ++i) se += expf(lg[base + i] - m);
    out[tid] = acc - m - logf(se);
}

// ---------------------------------------------------------------------------
// Host-side launcher
// ---------------------------------------------------------------------------
extern "C" void kernel_launch(void* const* d_in, const int* in_sizes, int n_in,
                              void* d_out, int out_size, void* d_ws, size_t ws_size,
                              hipStream_t stream)
{
    const float* x        = (const float*)d_in[0];
    const float* pos      = (const float*)d_in[1];
    const int*   ei       = (const int*)  d_in[2];
    const int*   batch    = (const int*)  d_in[3];
    const float* c1w      = (const float*)d_in[4];
    const float* c1b      = (const float*)d_in[5];
    const float* bn1g     = (const float*)d_in[6];
    const float* bn1b     = (const float*)d_in[7];
    const float* c2w      = (const float*)d_in[8];
    const float* c2b      = (const float*)d_in[9];
    const float* bn2g     = (const float*)d_in[10];
    const float* bn2b     = (const float*)d_in[11];
    const float* c3w      = (const float*)d_in[12];
    const float* c3b      = (const float*)d_in[13];
    const float* bn3g     = (const float*)d_in[14];
    const float* bn3b     = (const float*)d_in[15];
    const float* pw1      = (const float*)d_in[16];
    const float* pb1      = (const float*)d_in[17];
    const float* pw2      = (const float*)d_in[18];
    const float* pb2      = (const float*)d_in[19];
    const float* g1w      = (const float*)d_in[20];
    const float* g1b      = (const float*)d_in[21];
    const float* g2w      = (const float*)d_in[22];
    const float* g2b      = (const float*)d_in[23];
    const float* dw       = (const float*)d_in[24];
    const float* db       = (const float*)d_in[25];
    const int E = in_sizes[2] / 2;
    const int* esrc = ei;
    const int* edst = ei + E;

    char* ws = (char*)d_ws;
    // region layout (phase-reused)
    const size_t OFF_A  = 0;                       // 132 MiB
    const size_t OFF_B  = 138412032;               // 132 MiB
    const size_t OFF_C  = 276824064;               // 66 MiB
    const size_t OFF_ST = 346030080;               // 64 KiB stats

    unsigned short* h1      = (unsigned short*)(ws + OFF_A);              // 131.1 MB
    float*          pooled2 = (float*)         (ws + OFF_B);              // 130.0 MB
    unsigned short* h2      = (unsigned short*)(ws + OFF_C);              //  65.0 MB
    float*          pooled3 = (float*)         (ws + OFF_A);              //  62.9 MB (h1 dead)
    float*          pe      = (float*)         (ws + OFF_A + 67108864);
    unsigned short* hflat   = (unsigned short*)(ws + OFF_A + 70254592);   //  31.5 MB
    float*          g1      = (float*)         (ws + OFF_B);              // pooled2 dead
    float*          deg     = (float*)         (ws + OFF_B + 8388608);
    float*          agg     = (float*)         (ws + OFF_B + 16777216);
    unsigned short* h1g     = (unsigned short*)(ws + OFF_B + 25165824);
    float*          g2      = (float*)         (ws + OFF_B + 33554432);
    float*          agg2    = (float*)         (ws + OFF_B + 41943040);
    float*          h2g     = (float*)         (ws + OFF_B + 50331648);
    float*          pooledB = (float*)         (ws + OFF_B + 58720256);
    float*          cnt     = (float*)         (ws + OFF_B + 58818560);

    float* sums1 = (float*)(ws + OFF_ST + 0);
    float* sc1   = (float*)(ws + OFF_ST + 256);
    float* sh1   = (float*)(ws + OFF_ST + 512);
    float* sums2 = (float*)(ws + OFF_ST + 1024);
    float* sc2   = (float*)(ws + OFF_ST + 2048);
    float* sh2   = (float*)(ws + OFF_ST + 2560);
    float* sums3 = (float*)(ws + OFF_ST + 4096);
    float* sc3   = (float*)(ws + OFF_ST + 8192);
    float* sh3   = (float*)(ws + OFF_ST + 8704);

    // ---- stage 1: temporal CNN ----
    hipMemsetAsync(ws + OFF_ST, 0, 65536, stream);

    conv1_stats_k<<<NN * 16, 256, 0, stream>>>(x, c1w, c1b, sums1);
    finalize_bn_k<<<1, C1, 0, stream>>>(sums1, bn1g, bn1b, 1.f / (NN * (float)TT), C1, sc1, sh1);
    conv1_apply_k<<<(NN * C1 * TT) / 256, 256, 0, stream>>>(x, c1w, c1b, sc1, sh1, h1);

    conv_gemm_k<<<dim3(C2 / 32, 16, NN), 256, 0, stream>>>(
        h1, c2w, c2b, pooled2, sums2, C1, TT, T2, C2, C1 * 7, W2);
    finalize_bn_k<<<1, C2, 0, stream>>>(sums2, bn2g, bn2b, 1.f / (NN * (float)T2), C2, sc2, sh2);
    bn_relu_bf16_k<<<(NN * C2 * W2) / 256, 256, 0, stream>>>(pooled2, sc2, sh2, h2, W2, C2);

    conv_gemm_k<<<dim3(C3 / 32, 4, NN), 256, 0, stream>>>(
        h2, c3w, c3b, pooled3, sums3, C2, W2, T3, C3, C2 * 7, W3);
    finalize_bn_k<<<1, C3, 0, stream>>>(sums3, bn3g, bn3b, 1.f / (NN * (float)T3), C3, sc3, sh3);

    // ---- stage 2: position MLP + flatten ----
    pos_mlp_k<<<NN, 128, 0, stream>>>(pos, pw1, pb1, pw2, pb2, pe);
    build_flat_k<<<(NN * FLAT) / 256, 256, 0, stream>>>(pooled3, sc3, sh3, pe, hflat);

    // ---- stage 3: GCN layer 1 ----
    gemm_bf16_k<<<dim3(EMB / 64, NN / 64), 256, 0, stream>>>(hflat, g1w, g1, NN, EMB, FLAT);
    hipMemsetAsync(deg, 0, NN * sizeof(float), stream);
    hipMemsetAsync(agg, 0, (size_t)NN * EMB * sizeof(float), stream);
    deg_count_k<<<(E + 255) / 256, 256, 0, stream>>>(edst, deg, E);
    edge_scatter_k<<<E, 256, 0, stream>>>(g1, deg, esrc, edst, agg);
    gcn_combine_k<<<(NN * EMB) / 256, 256, 0, stream>>>(g1, agg, deg, g1b, nullptr, h1g);

    // ---- GCN layer 2 ----
    gemm_bf16_k<<<dim3(EMB / 64, NN / 64), 256, 0, stream>>>(h1g, g2w, g2, NN, EMB, EMB);
    hipMemsetAsync(agg2, 0, (size_t)NN * EMB * sizeof(float), stream);
    edge_scatter_k<<<E, 256, 0, stream>>>(g2, deg, esrc, edst, agg2);
    gcn_combine_k<<<(NN * EMB) / 256, 256, 0, stream>>>(g2, agg2, deg, g2b, h2g, nullptr);

    // ---- batch mean pool + dense + log-softmax ----
    hipMemsetAsync(pooledB, 0, (size_t)BB * EMB * sizeof(float), stream);
    hipMemsetAsync(cnt, 0, BB * sizeof(float), stream);
    batch_count_k<<<(NN + 255) / 256, 256, 0, stream>>>(batch, cnt, NN);
    batch_pool_k<<<(NN * EMB) / 256, 256, 0, stream>>>(h2g, batch, pooledB);
    final_dense_k<<<1, BB * NCLS, 0, stream>>>(pooledB, cnt, dw, db, (float*)d_out);
}
